// GateUpv_4_35828617183604
// MI455X (gfx1250) — matrix operand order repaired and checked by reading, not compiled, not run on hardware
//
#include <hip/hip_runtime.h>
#include <hip/hip_bf16.h>

// ---------------- WMMA fragment types (CDNA5 gfx1250, wave32) ----------------
typedef __attribute__((ext_vector_type(16))) __bf16        v16bf;
typedef __attribute__((ext_vector_type(8)))  float         v8f;
typedef __attribute__((ext_vector_type(4)))  unsigned int  v4u;

union ABFrag { v16bf v; v4u q[2]; };

__device__ __forceinline__ unsigned short f2bf(float f) {
  unsigned int u = __float_as_uint(f);
  u += 0x7FFFu + ((u >> 16) & 1u);        // round-to-nearest-even
  return (unsigned short)(u >> 16);
}

// ---------------- problem constants ----------------
#define BATCH   2
#define CCH     64          // channels of x1 / x2_
#define HO      128         // upsampled H
#define WWO     128         // upsampled W
#define NPIX    (BATCH * HO * WWO)   // 32768 pixels
#define KATT    16

// ---------------- workspace layout (byte offsets, 16B-aligned) ----------------
#define X1U_OFF  0UL                    // fp32 NCHW x1u                (8 MiB)
#define H_OFF    8388608UL              // bf16 [pix][128]: gate|x1u    (8 MiB)
#define KF_OFF   16777216UL             // fp32 [pix][16] Kf            (2 MiB)
#define W1B_OFF  18874368UL             // bf16 [64][1152]
#define W2B_OFF  19021824UL             // bf16 [64][576]
#define WKB_OFF  19095552UL             // bf16 [16][576]
#define Y1_OFF   19113984UL             // fp32 [pix][64] conv1 raw     (8 MiB)
#define H2_OFF   27502592UL             // bf16 [pix][64] bn1+relu      (4 MiB)
#define Y2_OFF   31696896UL             // fp32 [pix][64] conv2 raw     (8 MiB)
#define STAT_OFF 40085504UL             // fp32 sum1[64] ss1[64] sum2[64] ss2[64]
#define AFF_OFF  40086528UL             // fp32 scale1 shift1 scale2 shift2

// ---------------- zero BN stat accumulators ----------------
__global__ void k_zero(float* __restrict__ p, int n) {
  int i = blockIdx.x * blockDim.x + threadIdx.x;
  if (i < n) p[i] = 0.0f;
}

// ---------------- repack OIHW fp32 weights -> bf16 [N][tap*Cin+ci] ----------------
__global__ void k_prepw(const float* __restrict__ src, unsigned short* __restrict__ dst,
                        int Cout, int Cin) {
  int idx = blockIdx.x * blockDim.x + threadIdx.x;
  int Ktot = Cin * 9;
  if (idx >= Cout * Ktot) return;
  int n = idx / Ktot, k = idx % Ktot;
  int tap = k / Cin, ci = k % Cin;
  int ky = tap / 3, kx = tap % 3;
  dst[idx] = f2bf(src[((n * Cin + ci) * 3 + ky) * 3 + kx]);
}

// ---------------- bilinear 2x upsample (align_corners) ----------------
__global__ void k_upsample(const float* __restrict__ x1, float* __restrict__ x1u,
                           unsigned short* __restrict__ h) {
  int idx = blockIdx.x * blockDim.x + threadIdx.x;     // ((b*64+c)*128+y)*128+x
  if (idx >= BATCH * CCH * HO * WWO) return;
  int x = idx & 127, y = (idx >> 7) & 127, c = (idx >> 14) & 63, b = idx >> 20;
  float fy = y * (63.0f / 127.0f);
  float fx = x * (63.0f / 127.0f);
  int y0 = (int)floorf(fy), x0 = (int)floorf(fx);
  int y1i = min(y0 + 1, 63), x1i = min(x0 + 1, 63);
  float wy = fy - (float)y0, wx = fx - (float)x0;
  const float* base = x1 + (((long)(b * 64 + c)) << 12);
  float v00 = base[(y0 << 6) + x0],  v01 = base[(y0 << 6) + x1i];
  float v10 = base[(y1i << 6) + x0], v11 = base[(y1i << 6) + x1i];
  float top = v00 * (1.f - wx) + v01 * wx;
  float bot = v10 * (1.f - wx) + v11 * wx;
  float v = top * (1.f - wy) + bot * wy;
  x1u[idx] = v;
  int p = (b << 14) + (y << 7) + x;
  h[p * 128 + 64 + c] = f2bf(v);
}

// ---------------- implicit-GEMM 3x3 SAME conv via V_WMMA_F32_16X16X32_BF16 ----
// Weights staged in LDS once per block (all 8 waves / 4 N-tiles share them).
// A fragments: 2x global_load_b128 per k-step; B fragments: 2x ds b128 reads.
template <int NT, int CINK>
__global__ void k_conv_wmma(const unsigned short* __restrict__ src, int srcStride, int srcOff,
                            const unsigned short* __restrict__ wgt,
                            const float* __restrict__ bias, int doRelu,
                            float* __restrict__ out, int outStride,
                            float* __restrict__ sumBuf, float* __restrict__ sumsqBuf) {
  constexpr int Ktot   = CINK * 9;
  constexpr int WQUADS = (NT * 16 * Ktot) / 8;   // uint4 count of weight matrix

  extern __shared__ char smem_raw[];
  unsigned short* sw = (unsigned short*)smem_raw;

  // ---- cooperative weight stage: global(b128) -> LDS ----
  {
    const v4u* g = (const v4u*)wgt;
    v4u* s = (v4u*)sw;
    for (int i = threadIdx.x; i < WQUADS; i += 256) s[i] = g[i];
  }
  __syncthreads();

  const int lane  = threadIdx.x & 31;
  const int wave  = threadIdx.x >> 5;
  const int strip = blockIdx.x * 8 + wave;      // 16 consecutive pixels (same row: W=128)
  const int m     = lane & 15;
  const int khalf = lane >> 4;
  const int p     = strip * 16 + m;             // this lane's A-row pixel
  const int b     = p >> 14, y = (p >> 7) & 127, x = p & 127;
  const int n0    = lane & 15;                  // B/D column within 16-wide tile

  const v4u vzero = {0u, 0u, 0u, 0u};
  v8f acc[NT] = {};

#pragma unroll 1
  for (int tap = 0; tap < 9; ++tap) {
    const int ky = tap / 3 - 1, kx = tap % 3 - 1;
    const int yy = y + ky, xx = x + kx;
    const bool valid = ((unsigned)yy < 128u) & ((unsigned)xx < 128u);
    const unsigned short* abase =
        src + (long)(((b << 7) + yy) << 7) * srcStride + (long)xx * srcStride + srcOff;
#pragma unroll
    for (int kc = 0; kc < CINK; kc += 32) {
      // A fragment (16x32 bf16): lane half holds K 0-7/16-23 vs 8-15/24-31
      ABFrag a;
      const unsigned short* ab = abase + kc + khalf * 8;
      a.q[0] = valid ? *(const v4u*)(ab)      : vzero;   // K group +0
      a.q[1] = valid ? *(const v4u*)(ab + 16) : vzero;   // K group +16
      const int kb = tap * CINK + kc + khalf * 8;        // B: K 0-7/16-23 vs 8-15/24-31
#pragma unroll
      for (int t = 0; t < NT; ++t) {
        ABFrag w;
        const unsigned short* wb = sw + (t * 16 + n0) * Ktot + kb;
        w.q[0] = *(const v4u*)(wb);
        w.q[1] = *(const v4u*)(wb + 16);
        acc[t] = __builtin_amdgcn_wmma_f32_16x16x32_bf16(
            false, a.v, false, w.v, (short)0, acc[t], false, false);
      }
    }
  }

  // D layout: N = lane&15, M = r + 8*(lane>>4)
#pragma unroll
  for (int t = 0; t < NT; ++t) {
    const int chn = t * 16 + n0;
    const float bb = bias ? bias[chn] : 0.0f;
    float s = 0.0f, s2 = 0.0f;
#pragma unroll
    for (int r = 0; r < 8; ++r) {
      float v = acc[t][r] + bb;
      if (doRelu) v = fmaxf(v, 0.0f);
      const int pr = strip * 16 + khalf * 8 + r;
      out[(long)pr * outStride + chn] = v;
      s += v;
      s2 = fmaf(v, v, s2);
    }
    if (sumBuf) {
      atomicAdd(&sumBuf[chn], s);
      atomicAdd(&sumsqBuf[chn], s2);
    }
  }
}

// ---------------- fused depthwise-Q + softmax attention + V gate ----------------
__global__ void k_gate(const float* __restrict__ x1u, const float* __restrict__ x2,
                       const float* __restrict__ kf, const float* __restrict__ Wq,
                       const float* __restrict__ bq, const float* __restrict__ Wv,
                       const float* __restrict__ bv, unsigned short* __restrict__ h) {
  int idx = blockIdx.x * blockDim.x + threadIdx.x;   // p*64 + c
  if (idx >= NPIX * CCH) return;
  int c = idx & 63, p = idx >> 6;
  int x = p & 127, y = (p >> 7) & 127, b = p >> 14;
  const float* ch = x1u + (((long)(b * 64 + c)) << 14);
  float q = bq[c];
#pragma unroll
  for (int ky = 0; ky < 3; ++ky)
#pragma unroll
    for (int kx = 0; kx < 3; ++kx) {
      int yy = y + ky - 1, xx = x + kx - 1;
      float v = (((unsigned)yy < 128u) & ((unsigned)xx < 128u)) ? ch[(yy << 7) + xx] : 0.0f;
      q = fmaf(Wq[c * 9 + ky * 3 + kx], v, q);
    }
  q = fmaxf(q, 0.0f);
  const float* kfr = kf + p * 16;
  float s[KATT];
  float mx = -3.4e38f;
#pragma unroll
  for (int k = 0; k < KATT; ++k) { s[k] = q * kfr[k]; mx = fmaxf(mx, s[k]); }
  float x2v = x2[(((long)(b * 64 + c)) << 14) + (y << 7) + x];
  float num = 0.0f, den = 0.0f;
#pragma unroll
  for (int k = 0; k < KATT; ++k) {
    float e = __expf(s[k] - mx);
    float V = fmaxf(fmaf(x2v, Wv[k], bv[k]), 0.0f);
    num = fmaf(e, V, num);
    den += e;
  }
  h[p * 128 + c] = f2bf(num / den);
}

// ---------------- finalize BN stats -> affine scale/shift ----------------
__global__ void k_bnstats(const float* __restrict__ sum, const float* __restrict__ sumsq,
                          const float* __restrict__ g, const float* __restrict__ beta,
                          float* __restrict__ scale, float* __restrict__ shift) {
  int c = threadIdx.x;
  if (c < 64) {
    const float invN = 1.0f / (float)NPIX;
    float mean = sum[c] * invN;
    float var  = sumsq[c] * invN - mean * mean;
    float inv  = rsqrtf(var + 1e-5f);
    float sc   = g[c] * inv;
    scale[c] = sc;
    shift[c] = beta[c] - mean * sc;
  }
}

// ---------------- BN+ReLU -> bf16 (conv2 input) ----------------
__global__ void k_bnrelu_bf16(const float* __restrict__ yin, const float* __restrict__ scale,
                              const float* __restrict__ shift, unsigned short* __restrict__ hout) {
  int idx = blockIdx.x * blockDim.x + threadIdx.x;
  if (idx >= NPIX * 64) return;
  int c = idx & 63;
  hout[idx] = f2bf(fmaxf(fmaf(yin[idx], scale[c], shift[c]), 0.0f));
}

// ---------------- BN+ReLU + channel split -> d_out ----------------
__global__ void k_final(const float* __restrict__ yin, const float* __restrict__ scale,
                        const float* __restrict__ shift, float* __restrict__ out) {
  int idx = blockIdx.x * blockDim.x + threadIdx.x;
  if (idx >= NPIX * 64) return;
  int c = idx & 63, p = idx >> 6;
  float v = fmaxf(fmaf(yin[idx], scale[c], shift[c]), 0.0f);
  int b = p >> 14, rest = p & 16383;
  long o;
  if (c < 32) o = ((long)(b * 32 + c) << 14) + rest;                    // o1
  else        o = (1L << 20) + ((long)(b * 32 + (c - 32)) << 14) + rest; // o2
  out[o] = v;
}

// =============================================================================
extern "C" void kernel_launch(void* const* d_in, const int* in_sizes, int n_in,
                              void* d_out, int out_size, void* d_ws, size_t ws_size,
                              hipStream_t stream) {
  (void)in_sizes; (void)n_in; (void)out_size; (void)ws_size;
  const float* x1  = (const float*)d_in[0];
  /* d_in[1] = x1_ : unused by the reference */
  const float* x2_ = (const float*)d_in[2];
  const float* Wq  = (const float*)d_in[3];
  const float* bq  = (const float*)d_in[4];
  const float* Wk  = (const float*)d_in[5];
  const float* bk  = (const float*)d_in[6];
  const float* Wv  = (const float*)d_in[7];
  const float* bv  = (const float*)d_in[8];
  const float* W1  = (const float*)d_in[9];
  const float* g1  = (const float*)d_in[10];
  const float* b1  = (const float*)d_in[11];
  const float* W2  = (const float*)d_in[12];
  const float* g2  = (const float*)d_in[13];
  const float* b2  = (const float*)d_in[14];
  float* out = (float*)d_out;

  char* ws = (char*)d_ws;
  float*          x1u  = (float*)(ws + X1U_OFF);
  unsigned short* h    = (unsigned short*)(ws + H_OFF);
  float*          kf   = (float*)(ws + KF_OFF);
  unsigned short* w1b  = (unsigned short*)(ws + W1B_OFF);
  unsigned short* w2b  = (unsigned short*)(ws + W2B_OFF);
  unsigned short* wkb  = (unsigned short*)(ws + WKB_OFF);
  float*          y1   = (float*)(ws + Y1_OFF);
  unsigned short* h2   = (unsigned short*)(ws + H2_OFF);
  float*          y2   = (float*)(ws + Y2_OFF);
  float*          stat = (float*)(ws + STAT_OFF);   // sum1 ss1 sum2 ss2
  float*          aff  = (float*)(ws + AFF_OFF);    // scale1 shift1 scale2 shift2

  const int ELEMS = NPIX * 64;               // 2,097,152
  const int EB    = (ELEMS + 255) / 256;     // 8192 blocks
  const int CB    = NPIX / 16 / 8;           // 256 conv blocks (8 waves x 16-pixel strips)

  // LDS bytes for staged weights
  const int LDS_K1 = 16 * (64 * 9) * 2;      // Kf : 18432
  const int LDS_C1 = 64 * (128 * 9) * 2;     // conv1: 147456
  const int LDS_C2 = 64 * (64 * 9) * 2;      // conv2: 73728

  // 0. zero BN stat accumulators (re-zeroed every call: graph-replay safe)
  k_zero<<<1, 256, 0, stream>>>(stat, 256);

  // 1. repack weights to bf16 [N][K]
  k_prepw<<<(64 * 1152 + 255) / 256, 256, 0, stream>>>(W1, w1b, 64, 128);
  k_prepw<<<(64 * 576  + 255) / 256, 256, 0, stream>>>(W2, w2b, 64, 64);
  k_prepw<<<(16 * 576  + 255) / 256, 256, 0, stream>>>(Wk, wkb, 16, 64);

  // 2. bilinear upsample: x1u fp32 NCHW + bf16 upper half of h
  k_upsample<<<EB, 256, 0, stream>>>(x1, x1u, h);

  // 3. Kf = relu(conv3x3(x1u, Wk) + bk) via WMMA (N=16, K=576)
  k_conv_wmma<1, 64><<<CB, 256, LDS_K1, stream>>>(
      h, 128, 64, wkb, bk, 1, kf, 16, nullptr, nullptr);

  // 4. attention gate -> bf16 lower half of h
  k_gate<<<EB, 256, 0, stream>>>(x1u, x2_, kf, Wq, bq, Wv, bv, h);

  // 5. conv1: [pix][128] -> [pix][64] raw + BN stats (K=1152)
  k_conv_wmma<4, 128><<<CB, 256, LDS_C1, stream>>>(
      h, 128, 0, w1b, nullptr, 0, y1, 64, stat + 0, stat + 64);

  // 6. BN1 affine, then BN1+ReLU -> bf16 conv2 input
  k_bnstats<<<1, 64, 0, stream>>>(stat + 0, stat + 64, g1, b1, aff + 0, aff + 64);
  k_bnrelu_bf16<<<EB, 256, 0, stream>>>(y1, aff + 0, aff + 64, h2);

  // 7. conv2: [pix][64] -> [pix][64] raw + BN stats (K=576)
  k_conv_wmma<4, 64><<<CB, 256, LDS_C2, stream>>>(
      h2, 64, 0, w2b, nullptr, 0, y2, 64, stat + 128, stat + 192);

  // 8. BN2 affine, then BN2+ReLU + split write
  k_bnstats<<<1, 64, 0, stream>>>(stat + 128, stat + 192, g2, b2, aff + 128, aff + 192);
  k_final<<<EB, 256, 0, stream>>>(y2, aff + 128, aff + 192, out);
}
